// DotAttention_53386443489419
// MI455X (gfx1250) — matrix operand order repaired and checked by reading, not compiled, not run on hardware
//
#include <hip/hip_runtime.h>

typedef __attribute__((ext_vector_type(16))) _Float16 v16h;
typedef __attribute__((ext_vector_type(8)))  float    v8f;

#define DHEAD 128
#define BM 128
#define BN 64

__device__ __forceinline__ unsigned short f2h(float x) {
  union { _Float16 h; unsigned short u; } c;
  c.h = (_Float16)x;
  return c.u;
}

__device__ __forceinline__ void packstore4(unsigned short* p, float a, float b,
                                           float c, float d) {
  union { unsigned short us[4]; uint2 u; } w;
  w.us[0] = f2h(a); w.us[1] = f2h(b); w.us[2] = f2h(c); w.us[3] = f2h(d);
  *(uint2*)p = w.u;
}

__global__ __launch_bounds__(256, 2) void
fattn_kernel(const float* __restrict__ Q, const float* __restrict__ K,
             const float* __restrict__ V, const int* __restrict__ VL,
             float* __restrict__ O, int Lq, int Lk)
{
  // K-tile B-fragments: 16 frags (4 ntile x 4 kfrag) x 32 lanes x 16 halves
  __shared__ alignas(32) unsigned short kbuf[2][16 * 512];
  // V-tile B-fragments: 16 frags (2 kkfrag x 8 d-tiles) x 32 lanes x 16 halves
  __shared__ alignas(32) unsigned short vbuf[2][16 * 512];
  // per-wave P transpose scratch (16x64 f16, two A-fragments)
  __shared__ alignas(32) unsigned short pscr[8][1024];

  const int tid  = threadIdx.x;
  const int lane = tid & 31;
  const int wid  = tid >> 5;
  const int lmod = lane & 15;
  const int lgrp = lane >> 4;
  const int b     = blockIdx.y;
  const int qbase = blockIdx.x * BM;
  const int valid = VL[b];
  const int nkt   = (valid + BN - 1) / BN;   // uniform per block

  // ---- Load Q strip as f16 A-fragments (16x32 each, D=128 -> 4 frags) ----
  // A layout: lane l: M=l%16, group=l>>4; elem e=2v+s -> K=(v&3)*2+s+group*8+(v>>2)*16
  v16h qa[4];
  {
    const float* qrow = Q + ((size_t)b * Lq + qbase + wid * 16 + lmod) * DHEAD;
    #pragma unroll
    for (int kf = 0; kf < 4; ++kf) {
      #pragma unroll
      for (int v = 0; v < 8; ++v) {
        const int kk = (v & 3) * 2 + lgrp * 8 + (v >> 2) * 16 + kf * 32;
        qa[kf][2 * v]     = (_Float16)qrow[kk];
        qa[kf][2 * v + 1] = (_Float16)qrow[kk + 1];
      }
    }
  }

  // constant all-ones B fragment: rowsum(P) = P x Ones via WMMA
  v16h onesb;
  #pragma unroll
  for (int e = 0; e < 16; ++e) onesb[e] = (_Float16)1.0f;

  const float* kbase_p = K + (size_t)b * Lk * DHEAD;
  const float* vbase_p = V + (size_t)b * Lk * DHEAD;
  float kr[16];   // prefetch of K rows 0..31 only (rest streamed in store phase)

  auto load_k_regs = [&](int krow) {
    #pragma unroll
    for (int i = 0; i < 4; ++i) {
      const int j     = (tid >> 5) + i * 8;        // 0..31
      const int dbase = (tid & 31) * 4;            // 0..124
      const float4 f4 = *(const float4*)(kbase_p + (size_t)(krow + j) * DHEAD + dbase);
      kr[4*i+0] = f4.x; kr[4*i+1] = f4.y; kr[4*i+2] = f4.z; kr[4*i+3] = f4.w;
    }
  };

  // B layout (from ISA sparse-B pattern): lane = N%16 + 16*((K%16)/8), half = K%8 + 8*((K%32)/16)
  auto store_tile = [&](int buf, int krow) {
    #pragma unroll
    for (int i = 0; i < 8; ++i) {
      // K element (row j = key index -> N, col d -> K-dim)
      const int j     = (tid >> 5) + i * 8;        // 0..63
      const int dbase = (tid & 31) * 4;
      const int kgrp  = (dbase >> 3) & 1;
      const int klane = (j & 15) + 16 * kgrp;
      const int kh0   = (dbase & 7) + 8 * ((dbase >> 4) & 1); // {0,4,8,12}
      const int kfrag = (j >> 4) * 4 + (dbase >> 5);   // ntile*4 + kf : 0..15
      float k0, k1, k2, k3;
      if (i < 4) {                                 // from prefetch regs
        k0 = kr[4*i+0]; k1 = kr[4*i+1]; k2 = kr[4*i+2]; k3 = kr[4*i+3];
      } else {                                     // rows 32..63 streamed now
        const float4 f4 = *(const float4*)(kbase_p + (size_t)(krow + j) * DHEAD + dbase);
        k0 = f4.x; k1 = f4.y; k2 = f4.z; k3 = f4.w;
      }
      packstore4(&kbuf[buf][kfrag * 512 + klane * 16 + kh0], k0, k1, k2, k3);
      // V element (row j -> K-dim, col d -> N), streamed (coalesced across lanes)
      const int d  = tid & 127;                    // 0..127
      const int jb = (tid >> 7) * 4 + i * 8;       // 0,4,...,60
      const float x0 = vbase_p[(size_t)(krow + jb + 0) * DHEAD + d];
      const float x1 = vbase_p[(size_t)(krow + jb + 1) * DHEAD + d];
      const float x2 = vbase_p[(size_t)(krow + jb + 2) * DHEAD + d];
      const float x3 = vbase_p[(size_t)(krow + jb + 3) * DHEAD + d];
      const int vgrp  = (jb >> 3) & 1;
      const int vlane = (d & 15) + 16 * vgrp;
      const int vh0   = (jb & 7) + 8 * ((jb >> 4) & 1); // {0,4,8,12}
      const int vfrag = (jb >> 5) * 8 + (d >> 4);  // kkfrag*8 + dtile : 0..15
      packstore4(&vbuf[buf][vfrag * 512 + vlane * 16 + vh0], x0, x1, x2, x3);
    }
  };

  const v8f vzero = {0.f,0.f,0.f,0.f,0.f,0.f,0.f,0.f};
  v8f oacc[8];
  #pragma unroll
  for (int dt = 0; dt < 8; ++dt) oacc[dt] = vzero;
  float m_i[8], l_i[8];
  #pragma unroll
  for (int r = 0; r < 8; ++r) { m_i[r] = -__builtin_inff(); l_i[r] = 0.f; }

  // (1/sqrt(128)) * log2(e): exp2-domain softmax
  const float SCL = 1.4426950408889634f * 0.08838834764831845f;

  load_k_regs(0);
  store_tile(0, 0);
  __syncthreads();

  for (int kt = 0; kt < nkt; ++kt) {
    const int cur = kt & 1;
    const int kb0 = kt * BN;
    if (kt + 1 < nkt) load_k_regs(kb0 + BN);     // prefetch next K half-tile

    const unsigned short* kb = kbuf[cur];
    const unsigned short* vb = vbuf[cur];

    // ---- S = Q * K^T (16x64 per wave) ----
    v8f s[4];
    #pragma unroll
    for (int nt = 0; nt < 4; ++nt) {
      v8f acc = vzero;
      #pragma unroll
      for (int kf = 0; kf < 4; ++kf) {
        const v16h bk = *(const v16h*)(kb + (nt * 4 + kf) * 512 + lane * 16);
        acc = __builtin_amdgcn_wmma_f32_16x16x32_f16(false, qa[kf], false, bk,
                                                     (short)0, acc, false, false);
      }
      s[nt] = acc;
    }

    // ---- scale + length mask (column is fixed per lane in C layout) ----
    #pragma unroll
    for (int nt = 0; nt < 4; ++nt) {
      const int col = kb0 + nt * 16 + lmod;
      const bool ok = col < valid;
      #pragma unroll
      for (int r = 0; r < 8; ++r)
        s[nt][r] = ok ? s[nt][r] * SCL : -__builtin_inff();
    }

    // ---- online softmax: max via 16-lane butterfly, sum via WMMA below ----
    float corr[8];
    unsigned int pk01[8], pk23[8];   // f16 probabilities packed two-per-reg
    #pragma unroll
    for (int r = 0; r < 8; ++r) {
      float mx = fmaxf(fmaxf(s[0][r], s[1][r]), fmaxf(s[2][r], s[3][r]));
      mx = fmaxf(mx, __shfl_xor(mx, 1, 32));
      mx = fmaxf(mx, __shfl_xor(mx, 2, 32));
      mx = fmaxf(mx, __shfl_xor(mx, 4, 32));
      mx = fmaxf(mx, __shfl_xor(mx, 8, 32));
      const float mnew = fmaxf(m_i[r], mx);
      corr[r] = exp2f(m_i[r] - mnew);
      m_i[r] = mnew;
      pk01[r] = (unsigned)f2h(exp2f(s[0][r] - mnew)) |
                ((unsigned)f2h(exp2f(s[1][r] - mnew)) << 16);
      pk23[r] = (unsigned)f2h(exp2f(s[2][r] - mnew)) |
                ((unsigned)f2h(exp2f(s[3][r] - mnew)) << 16);
    }

    // ---- bounce P (16x64) through per-wave LDS into two A-fragments ----
    unsigned short* ps = pscr[wid];
    #pragma unroll
    for (int r = 0; r < 8; ++r) {
      const int tl   = 8 * lgrp + r + 16 * (lmod >> 3); // lane = M + 16*((K%16)/8)
      const int base = tl * 16 + (lmod & 7);            // half = K%8 + 8*((K%32)/16)
      ps[base]           = (unsigned short)(pk01[r] & 0xffffu);
      ps[base + 8]       = (unsigned short)(pk01[r] >> 16);
      ps[512 + base]     = (unsigned short)(pk23[r] & 0xffffu);
      ps[512 + base + 8] = (unsigned short)(pk23[r] >> 16);
    }
    asm volatile("s_wait_dscnt 0x0" ::: "memory");      // in-wave RAW, no barrier
    const v16h pa0 = *(const v16h*)(ps + lane * 16);
    const v16h pa1 = *(const v16h*)(ps + 512 + lane * 16);

    // ---- row sums: l_frag = P x Ones (replaces 32 bpermutes) ----
    v8f lsum = __builtin_amdgcn_wmma_f32_16x16x32_f16(false, pa0, false, onesb,
                                                      (short)0, vzero, false, false);
    lsum = __builtin_amdgcn_wmma_f32_16x16x32_f16(false, pa1, false, onesb,
                                                  (short)0, lsum, false, false);
    #pragma unroll
    for (int r = 0; r < 8; ++r)
      l_i[r] = l_i[r] * corr[r] + lsum[r];

    // ---- O = corr*O + P * V ----
    #pragma unroll
    for (int dt = 0; dt < 8; ++dt) {
      v8f o = oacc[dt];
      #pragma unroll
      for (int r = 0; r < 8; ++r) o[r] *= corr[r];
      const v16h bv0 = *(const v16h*)(vb + dt * 512 + lane * 16);
      o = __builtin_amdgcn_wmma_f32_16x16x32_f16(false, pa0, false, bv0,
                                                 (short)0, o, false, false);
      const v16h bv1 = *(const v16h*)(vb + (8 + dt) * 512 + lane * 16);
      o = __builtin_amdgcn_wmma_f32_16x16x32_f16(false, pa1, false, bv1,
                                                 (short)0, o, false, false);
      oacc[dt] = o;
    }

    if (kt + 1 < nkt) store_tile(cur ^ 1, kb0 + BN);  // fill other buffer
    __syncthreads();
  }

  // ---- epilogue: divide by l, write coalesced f32 ----
  float inv[8];
  #pragma unroll
  for (int r = 0; r < 8; ++r) inv[r] = 1.0f / l_i[r];
  float* orow = O + ((size_t)b * Lq + qbase + wid * 16) * DHEAD;
  #pragma unroll
  for (int dt = 0; dt < 8; ++dt) {
    #pragma unroll
    for (int r = 0; r < 8; ++r)
      orow[(size_t)(8 * lgrp + r) * DHEAD + dt * 16 + lmod] = oacc[dt][r] * inv[r];
  }
}

extern "C" void kernel_launch(void* const* d_in, const int* in_sizes, int n_in,
                              void* d_out, int out_size, void* d_ws, size_t ws_size,
                              hipStream_t stream) {
  const float* Q  = (const float*)d_in[0];
  const float* K  = (const float*)d_in[1];
  const float* V  = (const float*)d_in[2];
  const int*   VL = (const int*)d_in[3];
  float* O = (float*)d_out;

  const int B  = in_sizes[3];
  const int Lq = in_sizes[0] / (B * DHEAD);
  const int Lk = in_sizes[1] / (B * DHEAD);

  dim3 grid(Lq / BM, B);
  fattn_kernel<<<grid, 256, 0, stream>>>(Q, K, V, VL, O, Lq, Lk);
}
